// GINFeatures_71150428225937
// MI455X (gfx1250) — matrix operand order repaired and checked by reading, not compiled, not run on hardware
//
#include <hip/hip_runtime.h>
#include <math.h>

// ---------------------------------------------------------------------------
// Types for CDNA5 WMMA (wave32)
// ---------------------------------------------------------------------------
typedef __attribute__((ext_vector_type(16))) __bf16 v16bf;
typedef __attribute__((ext_vector_type(8)))  __bf16 v8bf;
typedef __attribute__((ext_vector_type(8)))  float  v8f;

static __device__ __forceinline__ unsigned short f32_to_bf16(float f) {
    unsigned int u = __float_as_uint(f);
    unsigned int rounding = 0x7FFFu + ((u >> 16) & 1u);
    return (unsigned short)((u + rounding) >> 16);
}

// ---------------------------------------------------------------------------
// Utility kernels
// ---------------------------------------------------------------------------
__global__ void zero_kernel(float* __restrict__ p, int n) {
    int i = blockIdx.x * blockDim.x + threadIdx.x;
    if (i < n) p[i] = 0.0f;
}

// agg[dst[e]] += h[src[e]]  (4 floats per thread)
__global__ void scatter_add_kernel(const float* __restrict__ h,
                                   const int* __restrict__ src,
                                   const int* __restrict__ dst,
                                   float* __restrict__ agg,
                                   int E, int dq /* = din/4 */) {
    int tid = blockIdx.x * blockDim.x + threadIdx.x;
    if (tid >= E * dq) return;
    int e = tid / dq;
    int f = (tid - e * dq) * 4;
    int din = dq * 4;
    int s = src[e];
    int d = dst[e];
    const float4 v = *(const float4*)(h + (size_t)s * din + f);
    float* a = agg + (size_t)d * din + f;
    atomicAdd(a + 0, v.x);
    atomicAdd(a + 1, v.y);
    atomicAdd(a + 2, v.z);
    atomicAdd(a + 3, v.w);
}

// zb = bf16(h + agg)
__global__ void add_convert_kernel(const float* __restrict__ h,
                                   const float* __restrict__ agg,
                                   unsigned short* __restrict__ zb, int n) {
    int i = blockIdx.x * blockDim.x + threadIdx.x;
    if (i < n) zb[i] = f32_to_bf16(h[i] + agg[i]);
}

// Wt[n][k] = bf16(W[k][n]);  W is K x Nout row-major
__global__ void wt_convert_kernel(const float* __restrict__ W,
                                  unsigned short* __restrict__ Wt,
                                  int K, int Nout) {
    int i = blockIdx.x * blockDim.x + threadIdx.x;
    if (i >= K * Nout) return;
    int k = i / Nout;
    int n = i - k * Nout;
    Wt[(size_t)n * K + k] = f32_to_bf16(W[i]);
}

// ---------------------------------------------------------------------------
// WMMA GEMM: C = relu(A(MxK,bf16) * B(KxN) + bias), B given transposed bf16.
// Block: 256 threads = 8 waves; block tile 128 rows x 64 cols.
// Wave w: rows [blockIdx.x*128 + 16w, +16), cols [blockIdx.y*64, +64) -> 4 accs.
// ---------------------------------------------------------------------------
__global__ __launch_bounds__(256) void gemm_bf16_relu(
    const unsigned short* __restrict__ Au,   // M x K row-major bf16
    const unsigned short* __restrict__ Btu,  // Nout x K row-major bf16 (B^T)
    const float* __restrict__ bias,          // Nout
    unsigned short* __restrict__ outb,       // bf16 out (or null)
    float* __restrict__ outf,                // f32 out  (or null)
    int M, int K, int Nout) {
    const __bf16* A  = (const __bf16*)Au;
    const __bf16* Bt = (const __bf16*)Btu;

    const int lane = threadIdx.x & 31;
    const int wave = threadIdx.x >> 5;
    const int lrow = lane & 15;   // A: row-in-tile / B,C: col-in-tile
    const int half = lane >> 4;   // half-wave select

    const int m0 = blockIdx.x * 128 + wave * 16;
    const int n0 = blockIdx.y * 64;

    int arow = m0 + lrow;
    if (arow >= M) arow = M - 1;  // clamp; tail rows masked at store
    const __bf16* Ap  = A + (size_t)arow * K;
    const __bf16* Bp0 = Bt + (size_t)(n0 +  0 + lrow) * K + 16 * half;
    const __bf16* Bp1 = Bt + (size_t)(n0 + 16 + lrow) * K + 16 * half;
    const __bf16* Bp2 = Bt + (size_t)(n0 + 32 + lrow) * K + 16 * half;
    const __bf16* Bp3 = Bt + (size_t)(n0 + 48 + lrow) * K + 16 * half;

    v8f acc0 = {}, acc1 = {}, acc2 = {}, acc3 = {};

    for (int kb = 0; kb < K; kb += 32) {
        // A fragment: half-wave holds 16 contiguous K
        // [16*half .. 16*half+15] (matches B)
        v8bf alo = *(const v8bf*)(Ap + kb + 16 * half);
        v8bf ahi = *(const v8bf*)(Ap + kb + 16 * half + 8);
        v16bf a = __builtin_shufflevector(alo, ahi,
                    0,1,2,3,4,5,6,7,8,9,10,11,12,13,14,15);
        // B fragments: lane = column, 16 contiguous K per half-wave
        v16bf b0 = *(const v16bf*)(Bp0 + kb);
        v16bf b1 = *(const v16bf*)(Bp1 + kb);
        v16bf b2 = *(const v16bf*)(Bp2 + kb);
        v16bf b3 = *(const v16bf*)(Bp3 + kb);

        acc0 = __builtin_amdgcn_wmma_f32_16x16x32_bf16(false, a, false, b0,
                                                       (short)0, acc0, false, false);
        acc1 = __builtin_amdgcn_wmma_f32_16x16x32_bf16(false, a, false, b1,
                                                       (short)0, acc1, false, false);
        acc2 = __builtin_amdgcn_wmma_f32_16x16x32_bf16(false, a, false, b2,
                                                       (short)0, acc2, false, false);
        acc3 = __builtin_amdgcn_wmma_f32_16x16x32_bf16(false, a, false, b3,
                                                       (short)0, acc3, false, false);
    }

    // Epilogue: C layout -> VGPR r holds M = 8*half + r, N = lane%16
    const int mrow = m0 + half * 8;
    #pragma unroll
    for (int r = 0; r < 8; ++r) {
        const int mm = mrow + r;
        if (mm >= M) continue;
        float vals[4] = { acc0[r], acc1[r], acc2[r], acc3[r] };
        #pragma unroll
        for (int t = 0; t < 4; ++t) {
            const int nn = n0 + t * 16 + lrow;
            float v = vals[t] + bias[nn];
            v = v > 0.0f ? v : 0.0f;
            const size_t idx = (size_t)mm * Nout + nn;
            if (outb) outb[idx] = f32_to_bf16(v);
            else      outf[idx] = v;
        }
    }
}

// ---------------------------------------------------------------------------
// BatchNorm (training mode, biased var) as partial sums + finalize + apply
// ---------------------------------------------------------------------------
#define BN_ROWS 40
__global__ __launch_bounds__(256) void bn_partial_kernel(
    const float* __restrict__ z, float* __restrict__ s1,
    float* __restrict__ s2, int M, int D) {
    int r0 = blockIdx.x * BN_ROWS;
    int rend = r0 + BN_ROWS;
    if (rend > M) rend = M;
    for (int c = threadIdx.x; c < D; c += 256) {
        float a = 0.0f, b = 0.0f;
        for (int r = r0; r < rend; ++r) {
            float v = z[(size_t)r * D + c];
            a += v;
            b += v * v;
        }
        atomicAdd(&s1[c], a);
        atomicAdd(&s2[c], b);
    }
}

__global__ void bn_finalize_kernel(const float* __restrict__ s1,
                                   const float* __restrict__ s2,
                                   const float* __restrict__ gamma,
                                   const float* __restrict__ beta,
                                   float* __restrict__ scale,
                                   float* __restrict__ shift, int M, int D) {
    int c = blockIdx.x * blockDim.x + threadIdx.x;
    if (c >= D) return;
    float inv = 1.0f / (float)M;
    float mu  = s1[c] * inv;
    float var = s2[c] * inv - mu * mu;
    float sc  = gamma[c] * rsqrtf(var + 1e-5f);
    scale[c] = sc;
    shift[c] = beta[c] - mu * sc;
}

__global__ void bn_apply_kernel(const float* __restrict__ z,
                                const float* __restrict__ scale,
                                const float* __restrict__ shift,
                                float* __restrict__ h, int n, int D) {
    int i = blockIdx.x * blockDim.x + threadIdx.x;
    if (i >= n) return;
    int c = i % D;
    h[i] = fmaf(z[i], scale[c], shift[c]);
}

// ---------------------------------------------------------------------------
// Global mean pool + final FC + tanh
// ---------------------------------------------------------------------------
__global__ void pool_sum_kernel(const float* __restrict__ h,
                                const int* __restrict__ batch,
                                float* __restrict__ psum, int N, int D) {
    int i = blockIdx.x * blockDim.x + threadIdx.x;
    if (i >= N * D) return;
    int r = i / D;
    int c = i - r * D;
    atomicAdd(&psum[(size_t)batch[r] * D + c], h[i]);
}

__global__ void pool_cnt_kernel(const int* __restrict__ batch,
                                float* __restrict__ pcnt, int N) {
    int i = blockIdx.x * blockDim.x + threadIdx.x;
    if (i < N) atomicAdd(&pcnt[batch[i]], 1.0f);
}

__global__ void final_fc_kernel(const float* __restrict__ psum,
                                const float* __restrict__ pcnt,
                                const float* __restrict__ W,
                                const float* __restrict__ b,
                                float* __restrict__ out, int G, int D) {
    int i = blockIdx.x * blockDim.x + threadIdx.x;
    if (i >= G * D) return;
    int g = i / D;
    int j = i - g * D;
    float cnt = pcnt[g];
    float inv = 1.0f / (cnt > 1.0f ? cnt : 1.0f);
    float acc = b[j];
    for (int k = 0; k < D; ++k)
        acc = fmaf(psum[(size_t)g * D + k] * inv, W[(size_t)k * D + j], acc);
    out[i] = tanhf(acc);
}

// ---------------------------------------------------------------------------
// Launch
// ---------------------------------------------------------------------------
extern "C" void kernel_launch(void* const* d_in, const int* in_sizes, int n_in,
                              void* d_out, int out_size, void* d_ws, size_t ws_size,
                              hipStream_t stream) {
    (void)in_sizes; (void)n_in; (void)out_size; (void)ws_size;
    const int N = 10000, E = 160000, G = 64, DIN = 128, D = 512, NL = 5;

    const float* x     = (const float*)d_in[0];
    const int*   ei    = (const int*)d_in[1];
    const int*   src   = ei;
    const int*   dst   = ei + E;
    const int*   batch = (const int*)d_in[2];

    // Workspace carve-out (256B aligned slabs)
    char* ws = (char*)d_ws;
    size_t off = 0;
    auto alloc = [&](size_t bytes) -> char* {
        char* p = ws + off;
        off += (bytes + 255) & ~(size_t)255;
        return p;
    };
    float*          h     = (float*)alloc((size_t)N * D * 4);
    float*          z     = (float*)alloc((size_t)N * D * 4);
    float*          agg   = (float*)alloc((size_t)N * D * 4);
    unsigned short* zb    = (unsigned short*)alloc((size_t)N * D * 2);
    unsigned short* zb2   = (unsigned short*)alloc((size_t)N * D * 2);
    unsigned short* wtb   = (unsigned short*)alloc((size_t)D * D * 2);
    float*          s1    = (float*)alloc(D * 4);
    float*          s2    = (float*)alloc(D * 4);
    float*          scale = (float*)alloc(D * 4);
    float*          shift = (float*)alloc(D * 4);
    float*          psum  = (float*)alloc((size_t)G * D * 4);
    float*          pcnt  = (float*)alloc(G * 4);

    const float* hin = x;
    int din = DIN;
    for (int l = 0; l < NL; ++l) {
        const float* W1 = (const float*)d_in[3 + 6 * l + 0];
        const float* b1 = (const float*)d_in[3 + 6 * l + 1];
        const float* W2 = (const float*)d_in[3 + 6 * l + 2];
        const float* b2 = (const float*)d_in[3 + 6 * l + 3];
        const float* gm = (const float*)d_in[3 + 6 * l + 4];
        const float* bt = (const float*)d_in[3 + 6 * l + 5];

        const int nAgg = N * din;
        zero_kernel<<<(nAgg + 255) / 256, 256, 0, stream>>>(agg, nAgg);

        const int dq  = din / 4;
        const int nSc = E * dq;
        scatter_add_kernel<<<(nSc + 255) / 256, 256, 0, stream>>>(hin, src, dst,
                                                                  agg, E, dq);
        add_convert_kernel<<<(nAgg + 255) / 256, 256, 0, stream>>>(hin, agg, zb,
                                                                   nAgg);
        const int nW1 = din * D;
        wt_convert_kernel<<<(nW1 + 255) / 256, 256, 0, stream>>>(W1, wtb, din, D);

        dim3 gg((N + 127) / 128, D / 64);
        // GEMM1: relu(z @ W1 + b1) -> bf16 (input of GEMM2)
        gemm_bf16_relu<<<gg, 256, 0, stream>>>(zb, wtb, b1, zb2, nullptr,
                                               N, din, D);
        const int nW2 = D * D;
        wt_convert_kernel<<<(nW2 + 255) / 256, 256, 0, stream>>>(W2, wtb, D, D);
        // GEMM2: relu(y1 @ W2 + b2) -> f32 (input of BN)
        gemm_bf16_relu<<<gg, 256, 0, stream>>>(zb2, wtb, b2, nullptr, z,
                                               N, D, D);

        zero_kernel<<<2, 256, 0, stream>>>(s1, D);
        zero_kernel<<<2, 256, 0, stream>>>(s2, D);
        const int nbBlocks = (N + BN_ROWS - 1) / BN_ROWS;
        bn_partial_kernel<<<nbBlocks, 256, 0, stream>>>(z, s1, s2, N, D);
        bn_finalize_kernel<<<2, 256, 0, stream>>>(s1, s2, gm, bt, scale, shift,
                                                  N, D);
        const int nz = N * D;
        bn_apply_kernel<<<(nz + 255) / 256, 256, 0, stream>>>(z, scale, shift,
                                                              h, nz, D);
        hin = h;
        din = D;
    }

    const float* Wfc = (const float*)d_in[3 + 6 * NL + 0];
    const float* bfc = (const float*)d_in[3 + 6 * NL + 1];

    zero_kernel<<<(G * D + 255) / 256, 256, 0, stream>>>(psum, G * D);
    zero_kernel<<<1, 256, 0, stream>>>(pcnt, G);
    const int nP = N * D;
    pool_sum_kernel<<<(nP + 255) / 256, 256, 0, stream>>>(h, batch, psum, N, D);
    pool_cnt_kernel<<<(N + 255) / 256, 256, 0, stream>>>(batch, pcnt, N);
    const int nO = G * D;
    final_fc_kernel<<<(nO + 255) / 256, 256, 0, stream>>>(psum, pcnt, Wfc, bfc,
                                                          (float*)d_out, G, D);
}
